// EncoderLayer_81690277970516
// MI455X (gfx1250) — matrix operand order repaired and checked by reading, not compiled, not run on hardware
//
#include <hip/hip_runtime.h>
#include <hip/hip_bf16.h>

// ---------------- problem constants (from reference) ----------------
#define BQ     8
#define LQ     4096
#define SQ     4096      // U_part == L (sampling cap hit)
#define DQ     45        // d_model
#define KP     64        // K padded to 2x32 for bf16 WMMA
#define UQ     45        // top-k query count
#define FH     128       // FFN hidden
#define NEG_INF (-3.402823466e+38f)

typedef __attribute__((ext_vector_type(16))) __bf16 v16bf;
typedef __attribute__((ext_vector_type(8)))  float  v8f;
typedef __attribute__((ext_vector_type(4)))  unsigned int u32x4;

union BfVec { v16bf v; u32x4 q[2]; };

__device__ __forceinline__ unsigned short f2bf(float f) {
    unsigned int u = __float_as_uint(f);
    return (unsigned short)((u + 0x7FFFu + ((u >> 16) & 1u)) >> 16);
}

// ---------------- 1) gather + cast to padded bf16 ----------------
// Xb[b,l,0:64] = bf16(x[b,l,k]) (k<45) else 0
// Kb[b,s,0:64] = bf16(x[b,idx[s],k]) (k<45) else 0
__global__ void __launch_bounds__(256)
gather_cast_kernel(const float* __restrict__ x, const int* __restrict__ idx,
                   unsigned short* __restrict__ Xb, unsigned short* __restrict__ Kb) {
    long t = (long)blockIdx.x * blockDim.x + threadIdx.x;
    const long nrow = (long)BQ * LQ;
    if (t >= 2 * nrow * KP) return;
    int  k = (int)(t & (KP - 1));
    long r = t >> 6;
    if (r < nrow) {
        unsigned short v = 0;
        if (k < DQ) v = f2bf(x[r * DQ + k]);
        Xb[r * KP + k] = v;
    } else {
        r -= nrow;
        int b = (int)(r / SQ);
        int s = (int)(r % SQ);
        int sl = idx[s];
        unsigned short v = 0;
        if (k < DQ) v = f2bf(x[((long)b * LQ + sl) * DQ + k]);
        Kb[r * KP + k] = v;
    }
}

// ---------------- 2) sampling GEMM: M[b,l] = max_s(QK) - sum_s(QK)/L ----------------
// Per wave: 16 query rows x all 4096 sampled keys, K=64 (2 x WMMA bf16 K=32).
__global__ void __launch_bounds__(256)
sampling_gemm_kernel(const unsigned short* __restrict__ Xb,
                     const unsigned short* __restrict__ Kb,
                     float* __restrict__ Mout) {
    const int blk   = blockIdx.x;            // BQ * (LQ/128)
    const int b     = blk / (LQ / 128);
    const int lblk  = blk % (LQ / 128);
    const int wave  = threadIdx.x >> 5;
    const int lane  = threadIdx.x & 31;
    const int lbase = lblk * 128 + wave * 16;
    const int mrow  = lane & 15;
    const int hi    = lane >> 4;             // 0: K 0-7/16-23, 1: K 8-15/24-31

    // A tile (16x64) in ISA layout, resident in registers
    const unsigned short* arow = Xb + ((size_t)((size_t)b * LQ + lbase + mrow)) * KP;
    BfVec A0, A1;
    A0.q[0] = *(const u32x4*)(arow + hi * 8);
    A0.q[1] = *(const u32x4*)(arow + hi * 8 + 16);
    A1.q[0] = *(const u32x4*)(arow + 32 + hi * 8);
    A1.q[1] = *(const u32x4*)(arow + 32 + hi * 8 + 16);

    float vmax[8], vsum[8];
#pragma unroll
    for (int i = 0; i < 8; ++i) { vmax[i] = NEG_INF; vsum[i] = 0.0f; }

    const unsigned short* kb = Kb + (size_t)b * SQ * KP;
#pragma unroll 2
    for (int st = 0; st < SQ; st += 16) {
        // B tile (64x16): lane n holds K = c*32 + hi*8 + 0..7 and c*32 + 16 + hi*8 + 0..7 of column st+n
        const unsigned short* brow = kb + (size_t)(st + mrow) * KP + hi * 8;
        BfVec B0, B1;
        B0.q[0] = *(const u32x4*)(brow);
        B0.q[1] = *(const u32x4*)(brow + 16);
        B1.q[0] = *(const u32x4*)(brow + 32);
        B1.q[1] = *(const u32x4*)(brow + 48);

        v8f acc = {};
        acc = __builtin_amdgcn_wmma_f32_16x16x32_bf16(false, A0.v, false, B0.v,
                                                      (short)0, acc, false, false);
        acc = __builtin_amdgcn_wmma_f32_16x16x32_bf16(false, A1.v, false, B1.v,
                                                      (short)0, acc, false, false);
#pragma unroll
        for (int i = 0; i < 8; ++i) {
            vmax[i] = fmaxf(vmax[i], acc[i]);
            vsum[i] += acc[i];
        }
    }

    // reduce across the 16 N-lanes of each half-wave (C layout: row = i + 8*hi)
#pragma unroll
    for (int i = 0; i < 8; ++i) {
#pragma unroll
        for (int off = 1; off < 16; off <<= 1) {
            vmax[i] = fmaxf(vmax[i], __shfl_xor(vmax[i], off, 32));
            vsum[i] += __shfl_xor(vsum[i], off, 32);
        }
    }
    if ((lane & 15) == 0) {
        const int rbase = lbase + hi * 8;
#pragma unroll
        for (int i = 0; i < 8; ++i)
            Mout[(size_t)b * LQ + rbase + i] = vmax[i] - vsum[i] * (1.0f / (float)LQ);
    }
}

// ---------------- 3) per-batch top-45 (iterative argmax, jax tie-break) ----------------
__global__ void __launch_bounds__(256)
topk_kernel(const float* __restrict__ M, int* __restrict__ Mtop) {
    const int b = blockIdx.x;
    const int tid = threadIdx.x;
    __shared__ float vals[LQ];
    __shared__ float rmax[256];
    __shared__ int   ridx[256];
    for (int i = tid; i < LQ; i += 256) vals[i] = M[(size_t)b * LQ + i];
    __syncthreads();
    for (int it = 0; it < UQ; ++it) {
        float best = NEG_INF; int bi = LQ;
        for (int i = tid; i < LQ; i += 256) {
            float v = vals[i];
            if (v > best || (v == best && i < bi)) { best = v; bi = i; }
        }
        rmax[tid] = best; ridx[tid] = bi;
        __syncthreads();
        for (int off = 128; off > 0; off >>= 1) {
            if (tid < off) {
                float v2 = rmax[tid + off]; int i2 = ridx[tid + off];
                if (v2 > rmax[tid] || (v2 == rmax[tid] && i2 < ridx[tid])) {
                    rmax[tid] = v2; ridx[tid] = i2;
                }
            }
            __syncthreads();
        }
        if (tid == 0) {
            Mtop[b * UQ + it] = ridx[0];
            vals[ridx[0]] = NEG_INF;
        }
        __syncthreads();
    }
}

// ---------------- 4) attention for the 45 selected queries ----------------
// one block per (b,u): scores row in LDS, softmax, weighted sum over d.
__global__ void __launch_bounds__(256)
attn_kernel(const float* __restrict__ x, const int* __restrict__ Mtop,
            float* __restrict__ attn_out) {
    const int b = blockIdx.x / UQ;
    const int u = blockIdx.x % UQ;
    const int tid = threadIdx.x;
    __shared__ float q[DQ];
    __shared__ float sc[LQ];
    __shared__ float red[256];

    const int qrow = Mtop[b * UQ + u];
    if (tid < DQ) q[tid] = x[((size_t)b * LQ + qrow) * DQ + tid];
    __syncthreads();

    const float scale = 1.0f / sqrtf((float)DQ);
    float lmax = NEG_INF;
    for (int l = tid; l < LQ; l += 256) {
        const float* xr = x + ((size_t)b * LQ + l) * DQ;
        float s = 0.0f;
#pragma unroll
        for (int k = 0; k < DQ; ++k) s += q[k] * xr[k];
        s *= scale;
        sc[l] = s;
        lmax = fmaxf(lmax, s);
    }
    red[tid] = lmax;
    __syncthreads();
    for (int off = 128; off > 0; off >>= 1) {
        if (tid < off) red[tid] = fmaxf(red[tid], red[tid + off]);
        __syncthreads();
    }
    const float m = red[0];
    __syncthreads();

    float lsum = 0.0f;
    for (int l = tid; l < LQ; l += 256) {
        float e = expf(sc[l] - m);
        sc[l] = e;
        lsum += e;
    }
    red[tid] = lsum;
    __syncthreads();
    for (int off = 128; off > 0; off >>= 1) {
        if (tid < off) red[tid] += red[tid + off];
        __syncthreads();
    }
    const float denom = red[0];
    __syncthreads();

    if (tid < DQ) {
        float acc = 0.0f;
        for (int l = 0; l < LQ; ++l)
            acc += sc[l] * x[((size_t)b * LQ + l) * DQ + tid];
        attn_out[((size_t)b * UQ + u) * DQ + tid] = acc / denom;
    }
}

// ---------------- 5) LN1 + FFN (over u-dim per reference) + LN2 ----------------
__global__ void __launch_bounds__(256)
ffn_kernel(const float* __restrict__ attn_out,
           const float* __restrict__ W1, const float* __restrict__ b1,
           const float* __restrict__ W2, const float* __restrict__ b2,
           const float* __restrict__ g1, const float* __restrict__ be1,
           const float* __restrict__ g2, const float* __restrict__ be2,
           float* __restrict__ out) {
    const int b = blockIdx.x;
    const int tid = threadIdx.x;
    __shared__ float h[UQ][DQ];     // LN1(attn_out), [u][d]
    __shared__ float gbuf[DQ][FH];  // relu features per d
    __shared__ float f2[UQ][DQ];    // pre-LN2, [u][d]

    if (tid < UQ) {
        const float* r = attn_out + ((size_t)b * UQ + tid) * DQ;
        float mean = 0.0f;
        for (int d = 0; d < DQ; ++d) mean += r[d];
        mean *= (1.0f / DQ);
        float var = 0.0f;
        for (int d = 0; d < DQ; ++d) { float t = r[d] - mean; var += t * t; }
        var *= (1.0f / DQ);
        float inv = 1.0f / sqrtf(var + 1e-12f);
        for (int d = 0; d < DQ; ++d)
            h[tid][d] = g1[d] * (r[d] - mean) * inv + be1[d];
    }
    __syncthreads();

    // g[d][hh] = relu(b1[hh] + sum_u h[u][d] * W1[hh,u])
    for (int i = tid; i < DQ * FH; i += 256) {
        int d = i >> 7, hh = i & (FH - 1);
        float acc = b1[hh];
        for (int uu = 0; uu < UQ; ++uu) acc += h[uu][d] * W1[hh * UQ + uu];
        gbuf[d][hh] = fmaxf(acc, 0.0f);
    }
    __syncthreads();

    // f2[u][d] = b2[u] + sum_hh g[d][hh] * W2[u,hh]
    for (int i = tid; i < UQ * DQ; i += 256) {
        int uu = i / DQ, d = i % DQ;
        float acc = b2[uu];
        for (int hh = 0; hh < FH; ++hh) acc += gbuf[d][hh] * W2[uu * FH + hh];
        f2[uu][d] = acc;
    }
    __syncthreads();

    if (tid < UQ) {
        float mean = 0.0f;
        for (int d = 0; d < DQ; ++d) mean += f2[tid][d];
        mean *= (1.0f / DQ);
        float var = 0.0f;
        for (int d = 0; d < DQ; ++d) { float t = f2[tid][d] - mean; var += t * t; }
        var *= (1.0f / DQ);
        float inv = 1.0f / sqrtf(var + 1e-12f);
        for (int d = 0; d < DQ; ++d)
            out[((size_t)b * UQ + tid) * DQ + d] =
                g2[d] * (f2[tid][d] - mean) * inv + be2[d];
    }
}

// ---------------- launch ----------------
extern "C" void kernel_launch(void* const* d_in, const int* in_sizes, int n_in,
                              void* d_out, int out_size, void* d_ws, size_t ws_size,
                              hipStream_t stream) {
    const float* x   = (const float*)d_in[0];
    const float* W1  = (const float*)d_in[1];
    const float* b1  = (const float*)d_in[2];
    const float* W2  = (const float*)d_in[3];
    const float* b2  = (const float*)d_in[4];
    const float* g1  = (const float*)d_in[5];
    const float* be1 = (const float*)d_in[6];
    const float* g2  = (const float*)d_in[7];
    const float* be2 = (const float*)d_in[8];
    const int*   idx = (const int*)d_in[9];
    float* out = (float*)d_out;

    // workspace layout
    const size_t XB_ELEMS = (size_t)BQ * LQ * KP;   // ushort
    unsigned short* Xb  = (unsigned short*)d_ws;
    unsigned short* Kb  = Xb + XB_ELEMS;
    float*          Mb  = (float*)(Kb + XB_ELEMS);
    int*            Mt  = (int*)(Mb + (size_t)BQ * LQ);
    float*          att = (float*)(Mt + BQ * 48);   // padded

    {
        long total = 2L * BQ * LQ * KP;
        int grid = (int)((total + 255) / 256);
        gather_cast_kernel<<<grid, 256, 0, stream>>>(x, idx, Xb, Kb);
    }
    sampling_gemm_kernel<<<BQ * (LQ / 128), 256, 0, stream>>>(Xb, Kb, Mb);
    topk_kernel<<<BQ, 256, 0, stream>>>(Mb, Mt);
    attn_kernel<<<BQ * UQ, 256, 0, stream>>>(x, Mt, att);
    ffn_kernel<<<BQ, 256, 0, stream>>>(att, W1, b1, W2, b2, g1, be1, g2, be2, out);
    (void)in_sizes; (void)n_in; (void)out_size; (void)ws_size;
}
